// GCN_Encoder_7627861917894
// MI455X (gfx1250) — matrix operand order repaired and checked by reading, not compiled, not run on hardware
//
#include <hip/hip_runtime.h>

typedef __attribute__((ext_vector_type(16))) __bf16 v16bf;
typedef __attribute__((ext_vector_type(8)))  float  v8f;

// ---------- helpers ----------
__device__ __forceinline__ unsigned short f2bf(float f) {
    unsigned u = __float_as_uint(f);
    unsigned r = u + 0x7FFFu + ((u >> 16) & 1u);   // round-to-nearest-even
    return (unsigned short)(r >> 16);
}

// ---------- degree / norm precompute ----------
__global__ void k_deg_init(float* deg, int n) {
    int i = blockIdx.x * blockDim.x + threadIdx.x;
    if (i < n) deg[i] = 1.0f;                       // self-loop weight
}

__global__ void k_deg_acc(const int* col, const float* w, float* deg, int e) {
    int i = blockIdx.x * blockDim.x + threadIdx.x;
    if (i < e) unsafeAtomicAdd(&deg[col[i]], w[i]);
}

__global__ void k_rsqrt_inplace(float* d, int n) {
    int i = blockIdx.x * blockDim.x + threadIdx.x;
    if (i < n) d[i] = rsqrtf(d[i]);
}

__global__ void k_norm(const int* row, const int* col, const float* w,
                       const float* dinv, float* nrm, int e) {
    int i = blockIdx.x * blockDim.x + threadIdx.x;
    if (i < e) nrm[i] = dinv[row[i]] * w[i] * dinv[col[i]];
}

// ---------- bf16 conversions ----------
__global__ void k_f32_to_bf16(const float* s, unsigned short* d, long n) {
    long i = (long)blockIdx.x * blockDim.x + threadIdx.x;
    if (i < n) d[i] = f2bf(s[i]);
}

// W[128][D] (row-major) -> Wt[D][128] bf16 (transposed, so K runs are contiguous)
__global__ void k_wT_bf16(const float* W, unsigned short* Wt, int D) {
    int i = blockIdx.x * blockDim.x + threadIdx.x;
    if (i < 128 * D) {
        int k = i / D, n = i % D;
        Wt[n * 128 + k] = f2bf(W[i]);
    }
}

// ---------- WMMA GEMM: C[M][Dout] = A[M][128](bf16) * Wt[Dout][128](bf16)^T ----------
__global__ void __launch_bounds__(256)
k_gemm_bf16(const unsigned short* __restrict__ A, const unsigned short* __restrict__ Bt,
            float* __restrict__ C, int M, int Dout) {
    // B staged in LDS; pitch 68 dwords (136 ushorts, 272B = 16*17) -> b128-aligned rows,
    // dword bank stride 68 % 64 = 4: lanes 0-15 touch all 64 banks once per phase.
    __shared__ unsigned sW[128 * 68];
    int tid = threadIdx.x;
    {
        const unsigned* Bw = (const unsigned*)Bt;
        int tot = Dout * 64;                        // dwords of Bt
        for (int t = tid; t < tot; t += 256) {
            int n = t >> 6, d = t & 63;
            sW[n * 68 + d] = Bw[n * 64 + d];
        }
    }
    __syncthreads();

    int lane = tid & 31, wave = tid >> 5;
    int lane16 = lane & 15, half = lane >> 4;
    int rowBase = (blockIdx.x * 8 + wave) * 16;
    int rowA = rowBase + lane16;
    if (rowA > M - 1) rowA = M - 1;                 // clamp loads; EXEC stays full

    // Preload A fragments (ISA §7.12.2): per lane, k-step kk is two contiguous
    // 16B runs at K = kk*32 + 8*half and K = kk*32 + 16 + 8*half.
    union Frag { v16bf v; uint4 q[2]; };
    Frag afr[4];
    const unsigned short* Arow = A + (long)rowA * 128;
#pragma unroll
    for (int kk = 0; kk < 4; ++kk) {
        const uint4* p = (const uint4*)(Arow + kk * 32 + 8 * half);
        afr[kk].q[0] = p[0];                        // K = base .. base+7
        afr[kk].q[1] = p[2];                        // K = base+16 .. base+23
    }

    bool fullWave = (rowBase + 16 <= M);            // wave-uniform tail check
    int ntiles = Dout >> 4;
    for (int nt = 0; nt < ntiles; ++nt) {
        int nb = nt * 16 + lane16;
        // Load ALL B fragments for this tile first (8x ds b128), then 4 WMMAs.
        Frag bfr[4];
#pragma unroll
        for (int kk = 0; kk < 4; ++kk) {
            const uint4* p = (const uint4*)&sW[nb * 68 + kk * 16 + 4 * half];
            bfr[kk].q[0] = p[0];                    // K = kk*32 + 8*half .. +7, then +16
            bfr[kk].q[1] = p[2];
        }
        v8f acc = {};
#pragma unroll
        for (int kk = 0; kk < 4; ++kk)
            acc = __builtin_amdgcn_wmma_f32_16x16x32_bf16(
                      false, afr[kk].v, false, bfr[kk].v, (short)0, acc, false, false);

        float* Cp = C + (long)rowBase * Dout + nt * 16 + lane16;
        if (fullWave) {
#pragma unroll
            for (int r = 0; r < 8; ++r)             // C/D: VGPR r -> M = r + 8*half
                Cp[(long)(r + 8 * half) * Dout] = acc[r];
        } else {
#pragma unroll
            for (int r = 0; r < 8; ++r) {
                int ro = rowBase + r + 8 * half;
                if (ro < M) Cp[(long)(r + 8 * half) * Dout] = acc[r];
            }
        }
    }
}

// ---------- self-loop + bias init: out = h * dinv^2 + b ----------
__global__ void k_selfinit(const float* h, const float* dinv, const float* b,
                           float* outp, int n, int D) {
    long idx = (long)blockIdx.x * blockDim.x + threadIdx.x;
    if (idx < (long)n * D) {
        int i = (int)(idx / D), c = (int)(idx % D);
        float di = dinv[i];
        outp[idx] = h[idx] * di * di + b[c];
    }
}

// ---------- edge scatter: acc[col] += h[row] * norm (L2-resident atomics) ----------
__global__ void k_scatter(const float* __restrict__ h, const float* __restrict__ nrm,
                          const int* __restrict__ row, const int* __restrict__ col,
                          float* acc, int e, int D) {
    long gid = (long)blockIdx.x * blockDim.x + threadIdx.x;
    int per = D >> 2;                                   // float4 chunks per edge
    long tot = (long)e * per;
    if (gid >= tot) return;
    int ei = (int)(gid / per);
    int c  = (int)(gid - (long)ei * per) * 4;
    float nm = nrm[ei];
    int r = row[ei], cl = col[ei];
    const float4 hv = *(const float4*)(h + (long)r * D + c);
    float* dst = acc + (long)cl * D + c;
    unsafeAtomicAdd(dst + 0, hv.x * nm);
    unsafeAtomicAdd(dst + 1, hv.y * nm);
    unsafeAtomicAdd(dst + 2, hv.z * nm);
    unsafeAtomicAdd(dst + 3, hv.w * nm);
}

// ---------- PReLU (per-channel slope) + bf16 cast, D=128 ----------
__global__ void k_prelu_bf16(const float* acc, const float* a,
                             unsigned short* outb, long n128) {
    long idx = (long)blockIdx.x * blockDim.x + threadIdx.x;
    if (idx < n128) {
        int c = (int)(idx & 127);
        float v = acc[idx];
        v = (v >= 0.0f) ? v : a[c] * v;
        outb[idx] = f2bf(v);
    }
}

// ---------- launch ----------
extern "C" void kernel_launch(void* const* d_in, const int* in_sizes, int n_in,
                              void* d_out, int out_size, void* d_ws, size_t ws_size,
                              hipStream_t stream) {
    const float* x  = (const float*)d_in[0];
    const int*   ei = (const int*)  d_in[1];
    const float* w  = (const float*)d_in[2];
    const float* W1 = (const float*)d_in[3];
    const float* b1 = (const float*)d_in[4];
    const float* a1 = (const float*)d_in[5];
    const float* W2 = (const float*)d_in[6];
    const float* b2 = (const float*)d_in[7];

    const int N = in_sizes[0] / 128;
    const int E = in_sizes[2];
    const int* row = ei;
    const int* col = ei + E;

    char* ws = (char*)d_ws;
    size_t o = 0;
    auto alloc = [&](size_t bytes) { size_t r = o; o += (bytes + 255) & ~(size_t)255; return r; };
    float*          dinv = (float*)(ws + alloc((size_t)N * 4));
    float*          nrm  = (float*)(ws + alloc((size_t)E * 4));
    unsigned short* xb   = (unsigned short*)(ws + alloc((size_t)N * 128 * 2)); // reused as h1b
    unsigned short* Wt1  = (unsigned short*)(ws + alloc((size_t)128 * 128 * 2));
    unsigned short* Wt2  = (unsigned short*)(ws + alloc((size_t)128 * 64 * 2));
    float*          h1   = (float*)(ws + alloc((size_t)N * 128 * 4));          // reused as h2
    float*          acc1 = (float*)(ws + alloc((size_t)N * 128 * 4));
    float*          out  = (float*)d_out;

    const int T = 256;
    auto blks = [&](long n) { return (int)((n + T - 1) / T); };

    // degrees & per-edge norm
    k_deg_init<<<blks(N), T, 0, stream>>>(dinv, N);
    k_deg_acc <<<blks(E), T, 0, stream>>>(col, w, dinv, E);
    k_rsqrt_inplace<<<blks(N), T, 0, stream>>>(dinv, N);
    k_norm    <<<blks(E), T, 0, stream>>>(row, col, w, dinv, nrm, E);

    // bf16 operands
    k_f32_to_bf16<<<blks((long)N * 128), T, 0, stream>>>(x, xb, (long)N * 128);
    k_wT_bf16<<<blks(128 * 128), T, 0, stream>>>(W1, Wt1, 128);
    k_wT_bf16<<<blks(128 * 64),  T, 0, stream>>>(W2, Wt2, 64);

    // ---- layer 1: GEMM -> self-loop+bias -> edge scatter -> PReLU(bf16) ----
    int gblk = (N + 127) / 128;
    k_gemm_bf16<<<gblk, 256, 0, stream>>>(xb, Wt1, h1, N, 128);
    k_selfinit <<<blks((long)N * 128), T, 0, stream>>>(h1, dinv, b1, acc1, N, 128);
    k_scatter  <<<blks((long)E * 32), T, 0, stream>>>(h1, nrm, row, col, acc1, E, 128);
    k_prelu_bf16<<<blks((long)N * 128), T, 0, stream>>>(acc1, a1, xb /*h1b*/, (long)N * 128);

    // ---- layer 2: GEMM -> self-loop+bias -> edge scatter (into d_out) ----
    k_gemm_bf16<<<gblk, 256, 0, stream>>>(xb /*h1b*/, Wt2, h1 /*h2*/, N, 64);
    k_selfinit <<<blks((long)N * 64), T, 0, stream>>>(h1 /*h2*/, dinv, b2, out, N, 64);
    k_scatter  <<<blks((long)E * 16), T, 0, stream>>>(h1 /*h2*/, nrm, row, col, out, E, 64);
}
